// ED_PointNeXt_89601607729773
// MI455X (gfx1250) — matrix operand order repaired and checked by reading, not compiled, not run on hardware
//
#include <hip/hip_runtime.h>
#include <hip/hip_bf16.h>
#include <math.h>

typedef __attribute__((ext_vector_type(16))) _Float16 v16h;
typedef __attribute__((ext_vector_type(8)))  _Float16 v8h;
typedef __attribute__((ext_vector_type(8)))  float    v8f;

#define BB    8
#define NN    4096
#define MM    1366          // ceil(4096/3)
#define RAD_INV 10.0f       // 1/RADIUS
#define NS    32            // NSAMPLE
#define EDK   10            // ED_NSAMPLE
#define ROWS_SA (BB*MM*NS)  // 349696  (multiple of 32)
#define ROWS_PT (BB*NN)     // 32768

static __device__ __forceinline__ v16h cat16(v8h lo, v8h hi) {
    return __builtin_shufflevector(lo, hi, 0,1,2,3,4,5,6,7,8,9,10,11,12,13,14,15);
}
static __device__ __forceinline__ v8f wmma_step(v16h a, v16h w, v8f c) {
    return __builtin_amdgcn_wmma_f32_16x16x32_f16(false, a, false, w, (short)0, c, false, false);
}

// ---------------------------------------------------------------- init: copy data to out, zero fsub + zbuf
__global__ void initk(const float* __restrict__ data, float* __restrict__ out,
                      float* __restrict__ fsub, unsigned int* __restrict__ zbuf_u32) {
    int t = blockIdx.x * blockDim.x + threadIdx.x;
    if (t < BB*NN*3) out[t] = data[t];
    if (t < BB*MM*128) fsub[t] = 0.0f;
    if (t < ROWS_PT*96/2) zbuf_u32[t] = 0u;       // zbuf: 32768 x 96 halfs
}

// ---------------------------------------------------------------- weight prep: f16, transposed [c][Kpad], zero-padded
// Wsa_t is K-PERMUTED to match SA A-layout: k=0..63 -> f0 feats (orig rows 3..66), k=64..66 -> dp (orig rows 0..2)
__global__ void prep_w(const float* __restrict__ Wsa, const float* __restrict__ Wfp,
                       const float* __restrict__ Wfin,
                       _Float16* __restrict__ wsa_t, _Float16* __restrict__ wfp_t,
                       _Float16* __restrict__ wfin_t) {
    int t = blockIdx.x * blockDim.x + threadIdx.x;
    if (t < 128*96) {
        int c = t / 96, k = t % 96;
        float v = (k < 64) ? Wsa[(k+3)*128 + c] : ((k < 67) ? Wsa[(k-64)*128 + c] : 0.0f);
        wsa_t[t] = (_Float16)v;
    }
    if (t < 64*192) {
        int c = t / 192, k = t % 192;
        wfp_t[t] = (_Float16)Wfp[k*64 + c];
    }
    if (t < 64*96) {
        int c = t / 96, k = t % 96;
        wfin_t[t] = (_Float16)((k < 68) ? Wfin[k*64 + c] : 0.0f);
    }
}

// ---------------------------------------------------------------- stem: f0 = relu(data @ W_stem + b), f16 twice
__global__ void stemk(const float* __restrict__ data, const float* __restrict__ W,
                      const float* __restrict__ bias, _Float16* __restrict__ f0h,
                      _Float16* __restrict__ afp) {
    int t = blockIdx.x * blockDim.x + threadIdx.x;
    if (t >= ROWS_PT * 64) return;
    int p = t >> 6, c = t & 63;
    float acc = bias[c] + data[p*3+0]*W[c] + data[p*3+1]*W[64+c] + data[p*3+2]*W[128+c];
    _Float16 h = (_Float16)fmaxf(acc, 0.0f);
    f0h[t] = h;
    afp[(size_t)p*192 + 128 + c] = h;   // FP-GEMM A cols 128..191
}

// ---------------------------------------------------------------- ball query: first 32 in-radius indices
__global__ void ballq(const float* __restrict__ data, int* __restrict__ idx) {
    int t = blockIdx.x * blockDim.x + threadIdx.x;
    if (t >= BB*MM) return;
    int b = t / MM, m = t % MM;
    const float* base = data + (size_t)b * NN * 3;
    float cx = base[m*9+0], cy = base[m*9+1], cz = base[m*9+2];
    int* o = idx + (size_t)t * NS;
    int cnt = 0, first = 0;
    for (int q = 0; q < NN && cnt < NS; ++q) {
        float dx = base[q*3+0]-cx, dy = base[q*3+1]-cy, dz = base[q*3+2]-cz;
        if (dx*dx + dy*dy + dz*dz < 0.01f) {
            if (cnt == 0) first = q;
            o[cnt++] = q;
        }
    }
    for (; cnt < NS; ++cnt) o[cnt] = first;
}

// ---------------------------------------------------------------- SA GEMM (WMMA f16) fused gather + max-pool
// rows = B*M*32, Kpad = 96 (permuted: f0[0..63], dp[64..66], 0[67..95]), cols = 128.
__global__ __launch_bounds__(32) void sa_gemm(const float* __restrict__ data,
                                              const _Float16* __restrict__ f0h,
                                              const int*   __restrict__ idx,
                                              const _Float16* __restrict__ wsa_t,
                                              const float* __restrict__ bsa,
                                              int* __restrict__ fsub_i) {
    const int ROWT = ROWS_SA / 16;                 // 21856
    int tile = blockIdx.x;
    int rt = tile % ROWT, ct = tile / ROWT;
    int row0 = rt * 16, c0 = ct * 16;
    int lane = threadIdx.x;
    int n = lane & 15, half = lane >> 4;

    int r   = row0 + n;
    int b   = r / (MM * NS);
    int rem = r - b * (MM * NS);
    int m   = rem >> 5;
    int s   = rem & 31;
    int j   = idx[((size_t)(b * MM + m)) * NS + s];
    const float* base = data + (size_t)b * NN * 3;
    _Float16 dp0 = (_Float16)((base[j*3+0] - base[m*9+0]) * RAD_INV);
    _Float16 dp1 = (_Float16)((base[j*3+1] - base[m*9+1]) * RAD_INV);
    _Float16 dp2 = (_Float16)((base[j*3+2] - base[m*9+2]) * RAD_INV);
    const v8h* arow = (const v8h*)(f0h + ((size_t)b * NN + j) * 64);   // 8 chunks
    const v8h* wrow = (const v8h*)(wsa_t + (size_t)(c0 + n) * 96);     // 12 chunks

    v8f acc = {};
    // k-step 0 (k=0..31): pure contiguous loads
    acc = wmma_step(cat16(arow[2*half], arow[2*half+1]),
                    cat16(wrow[2*half], wrow[2*half+1]), acc);
    // k-step 1 (k=32..63)
    acc = wmma_step(cat16(arow[4+2*half], arow[5+2*half]),
                    cat16(wrow[4+2*half], wrow[5+2*half]), acc);
    // k-step 2 (k=64..95): A built in registers (dp + zeros), W rows are pre-zero-padded
    {
        _Float16 z = (_Float16)0.0f;
        v8h lo = {};
        lo[0] = (half == 0) ? dp0 : z;
        lo[1] = (half == 0) ? dp1 : z;
        lo[2] = (half == 0) ? dp2 : z;
        v8h hi = {};
        acc = wmma_step(cat16(lo, hi),
                        cat16(wrow[8+2*half], wrow[9+2*half]), acc);
    }
    // bias + relu + max over the 16 tile rows (all rows share (b,m))
    float bias = bsa[c0 + n];
    float mx = 0.0f;
    #pragma unroll
    for (int i = 0; i < 8; ++i) mx = fmaxf(mx, fmaxf(acc[i] + bias, 0.0f));
    mx = fmaxf(mx, __shfl_xor(mx, 16, 32));
    if (half == 0) {
        // relu output >= 0 and fsub zero-init => int-pattern max == float max
        atomicMax(&fsub_i[((size_t)(b * MM + m)) * 128 + c0 + n], __float_as_int(mx));
    }
}

// ---------------------------------------------------------------- 3-NN interpolation -> afp cols 0..127 (f16)
__global__ void knn3(const float* __restrict__ data, const float* __restrict__ fsub,
                     _Float16* __restrict__ afp) {
    int t = blockIdx.x * blockDim.x + threadIdx.x;
    if (t >= ROWS_PT) return;
    int b = t >> 12, n = t & (NN-1);
    const float* base = data + (size_t)b * NN * 3;
    float px = base[n*3+0], py = base[n*3+1], pz = base[n*3+2];
    float d0 = 3.4e38f, d1 = 3.4e38f, d2 = 3.4e38f;
    int i0 = 0, i1 = 0, i2 = 0;
    for (int m = 0; m < MM; ++m) {
        float dx = base[m*9+0]-px, dy = base[m*9+1]-py, dz = base[m*9+2]-pz;
        float d = dx*dx + dy*dy + dz*dz;
        if (d < d0)      { d2=d1;i2=i1; d1=d0;i1=i0; d0=d;i0=m; }
        else if (d < d1) { d2=d1;i2=i1; d1=d;i1=m; }
        else if (d < d2) { d2=d;i2=m; }
    }
    float w0 = 1.0f/(d0+1e-8f), w1 = 1.0f/(d1+1e-8f), w2 = 1.0f/(d2+1e-8f);
    float wsum = w0 + w1 + w2;
    w0 /= wsum; w1 /= wsum; w2 /= wsum;
    const float* r0 = fsub + ((size_t)b*MM + i0) * 128;
    const float* r1 = fsub + ((size_t)b*MM + i1) * 128;
    const float* r2 = fsub + ((size_t)b*MM + i2) * 128;
    _Float16* o = afp + (size_t)t * 192;
    for (int c = 0; c < 128; ++c)
        o[c] = (_Float16)(w0*r0[c] + w1*r1[c] + w2*r2[c]);
}

// ---------------------------------------------------------------- FP GEMM (WMMA): rows 32768, K=192, cols 64
__global__ __launch_bounds__(32) void fp_gemm(const _Float16* __restrict__ afp,
                                              const _Float16* __restrict__ wfp_t,
                                              const float* __restrict__ bfp,
                                              _Float16* __restrict__ zbuf) {
    const int ROWT = ROWS_PT / 16;                 // 2048
    int tile = blockIdx.x;
    int rt = tile % ROWT, ct = tile / ROWT;
    int row0 = rt * 16, c0 = ct * 16;
    int lane = threadIdx.x;
    int n = lane & 15, half = lane >> 4;
    int r = row0 + n;
    const v8h* arow = (const v8h*)(afp + (size_t)r * 192);             // 24 chunks
    const v8h* wrow = (const v8h*)(wfp_t + (size_t)(c0 + n) * 192);    // 24 chunks

    v8f acc = {};
    #pragma unroll
    for (int ks = 0; ks < 6; ++ks) {
        v16h a = cat16(arow[4*ks + 2*half], arow[4*ks + 2*half + 1]);
        v16h w = cat16(wrow[4*ks + 2*half], wrow[4*ks + 2*half + 1]);
        acc = wmma_step(a, w, acc);
    }
    float bias = bfp[c0 + n];
    #pragma unroll
    for (int i = 0; i < 8; ++i) {
        int rr = row0 + i + 8 * half;
        zbuf[(size_t)rr * 96 + c0 + n] = (_Float16)fmaxf(acc[i] + bias, 0.0f);
    }
}

// ---------------------------------------------------------------- ED branch: 10-NN, 3x3 cov, eigvals, 3->4->4 MLP
__global__ void edk_kernel(const float* __restrict__ data,
                           const float* __restrict__ We1, const float* __restrict__ be1,
                           const float* __restrict__ We2, const float* __restrict__ be2,
                           _Float16* __restrict__ zbuf) {
    int t = blockIdx.x * blockDim.x + threadIdx.x;
    if (t >= ROWS_PT) return;
    int b = t >> 12, n = t & (NN-1);
    const float* base = data + (size_t)b * NN * 3;
    float px = base[n*3+0], py = base[n*3+1], pz = base[n*3+2];
    float bd[EDK]; int bi[EDK];
    for (int i = 0; i < EDK; ++i) { bd[i] = 3.4e38f; bi[i] = 0; }
    for (int q = 0; q < NN; ++q) {
        float dx = base[q*3+0]-px, dy = base[q*3+1]-py, dz = base[q*3+2]-pz;
        float d = dx*dx + dy*dy + dz*dz;
        if (d < bd[EDK-1]) {
            int pos = EDK-1;
            while (pos > 0 && d < bd[pos-1]) { bd[pos]=bd[pos-1]; bi[pos]=bi[pos-1]; --pos; }
            bd[pos] = d; bi[pos] = q;
        }
    }
    float nx[EDK], ny[EDK], nz[EDK];
    float mx = 0, my = 0, mz = 0;
    for (int i = 0; i < EDK; ++i) {
        int q = bi[i];
        nx[i] = base[q*3+0]; ny[i] = base[q*3+1]; nz[i] = base[q*3+2];
        mx += nx[i]; my += ny[i]; mz += nz[i];
    }
    mx *= 0.1f; my *= 0.1f; mz *= 0.1f;
    float cxx=0,cxy=0,cxz=0,cyy=0,cyz=0,czz=0;
    for (int i = 0; i < EDK; ++i) {
        float ax = nx[i]-mx, ay = ny[i]-my, az = nz[i]-mz;
        cxx += ax*ax; cxy += ax*ay; cxz += ax*az;
        cyy += ay*ay; cyz += ay*az; czz += az*az;
    }
    cxx *= 0.1f; cxy *= 0.1f; cxz *= 0.1f; cyy *= 0.1f; cyz *= 0.1f; czz *= 0.1f;
    float qd = (cxx + cyy + czz) * (1.0f/3.0f);
    float p1 = cxy*cxy + cxz*cxz + cyz*cyz;
    float p2 = (cxx-qd)*(cxx-qd) + (cyy-qd)*(cyy-qd) + (czz-qd)*(czz-qd) + 2.0f*p1;
    float e0, e1, e2;
    if (p2 <= 1e-30f) { e0 = e1 = e2 = qd; }
    else {
        float p  = sqrtf(p2 * (1.0f/6.0f));
        float iv = 1.0f / p;
        float b00 = (cxx-qd)*iv, b11 = (cyy-qd)*iv, b22 = (czz-qd)*iv;
        float b01 = cxy*iv, b02 = cxz*iv, b12 = cyz*iv;
        float detB = b00*(b11*b22 - b12*b12) - b01*(b01*b22 - b12*b02) + b02*(b01*b12 - b11*b02);
        float rr = fminf(1.0f, fmaxf(-1.0f, 0.5f * detB));
        float phi = acosf(rr) * (1.0f/3.0f);
        float big   = qd + 2.0f*p*cosf(phi);
        float small = qd + 2.0f*p*cosf(phi + 2.0943951023931953f);
        float mid   = 3.0f*qd - big - small;
        e0 = small; e1 = mid; e2 = big;
    }
    float h[4];
    #pragma unroll
    for (int jj = 0; jj < 4; ++jj)
        h[jj] = fmaxf(e0*We1[jj] + e1*We1[4+jj] + e2*We1[8+jj] + be1[jj], 0.0f);
    #pragma unroll
    for (int jj = 0; jj < 4; ++jj) {
        float v = be2[jj];
        #pragma unroll
        for (int i = 0; i < 4; ++i) v += h[i] * We2[i*4+jj];
        zbuf[(size_t)t * 96 + 64 + jj] = (_Float16)v;
    }
}

// ---------------------------------------------------------------- FIN GEMM (WMMA): rows 32768, Kpad=96, cols 64
__global__ __launch_bounds__(32) void fin_gemm(const _Float16* __restrict__ zbuf,
                                               const _Float16* __restrict__ wfin_t,
                                               const float* __restrict__ bfin,
                                               float* __restrict__ zc) {
    const int ROWT = ROWS_PT / 16;
    int tile = blockIdx.x;
    int rt = tile % ROWT, ct = tile / ROWT;
    int row0 = rt * 16, c0 = ct * 16;
    int lane = threadIdx.x;
    int n = lane & 15, half = lane >> 4;
    int r = row0 + n;
    const v8h* arow = (const v8h*)(zbuf + (size_t)r * 96);             // 12 chunks
    const v8h* wrow = (const v8h*)(wfin_t + (size_t)(c0 + n) * 96);    // 12 chunks

    v8f acc = {};
    #pragma unroll
    for (int ks = 0; ks < 3; ++ks) {
        v16h a = cat16(arow[4*ks + 2*half], arow[4*ks + 2*half + 1]);
        v16h w = cat16(wrow[4*ks + 2*half], wrow[4*ks + 2*half + 1]);
        acc = wmma_step(a, w, acc);
    }
    float bias = bfin[c0 + n];
    #pragma unroll
    for (int i = 0; i < 8; ++i) {
        int rr = row0 + i + 8 * half;
        zc[(size_t)rr * 64 + c0 + n] = acc[i] + bias;
    }
}

// ---------------------------------------------------------------- per-channel batchnorm stats (1 block / channel)
__global__ void bnstats(const float* __restrict__ zc, float* __restrict__ stats) {
    __shared__ float ss[256], ss2[256];
    int c = blockIdx.x, tid = threadIdx.x;
    float s = 0.0f, s2 = 0.0f;
    for (int r = tid; r < ROWS_PT; r += 256) {
        float v = zc[(size_t)r * 64 + c];
        s += v; s2 += v * v;
    }
    ss[tid] = s; ss2[tid] = s2;
    __syncthreads();
    for (int off = 128; off > 0; off >>= 1) {
        if (tid < off) { ss[tid] += ss[tid+off]; ss2[tid] += ss2[tid+off]; }
        __syncthreads();
    }
    if (tid == 0) {
        float mu  = ss[0]  * (1.0f / ROWS_PT);
        float var = ss2[0] * (1.0f / ROWS_PT) - mu * mu;
        stats[c]      = mu;
        stats[64 + c] = rsqrtf(var + 1e-5f);
    }
}

// ---------------------------------------------------------------- normalize + relu + transpose to (B,64,N)
__global__ void finalize(const float* __restrict__ zc, const float* __restrict__ stats,
                         const float* __restrict__ gamma, const float* __restrict__ beta,
                         float* __restrict__ outz) {
    int t = blockIdx.x * blockDim.x + threadIdx.x;
    if (t >= ROWS_PT * 64) return;
    int r = t >> 6, c = t & 63;
    int b = r >> 12, n = r & (NN-1);
    float v = (zc[t] - stats[c]) * stats[64+c] * gamma[c] + beta[c];
    v = fmaxf(v, 0.0f);
    outz[((size_t)b * 64 + c) * NN + n] = v;
}

// ================================================================ launch
extern "C" void kernel_launch(void* const* d_in, const int* in_sizes, int n_in,
                              void* d_out, int out_size, void* d_ws, size_t ws_size,
                              hipStream_t stream) {
    const float* data   = (const float*)d_in[0];
    const float* W_stem = (const float*)d_in[2];
    const float* b_stem = (const float*)d_in[3];
    const float* W_sa   = (const float*)d_in[4];
    const float* b_sa   = (const float*)d_in[5];
    const float* W_fp   = (const float*)d_in[6];
    const float* b_fp   = (const float*)d_in[7];
    const float* W_e1   = (const float*)d_in[8];
    const float* b_e1   = (const float*)d_in[9];
    const float* W_e2   = (const float*)d_in[10];
    const float* b_e2   = (const float*)d_in[11];
    const float* W_fin  = (const float*)d_in[12];
    const float* b_fin  = (const float*)d_in[13];
    const float* gamma  = (const float*)d_in[14];
    const float* beta   = (const float*)d_in[15];

    char* ws = (char*)d_ws;
    _Float16* f0h    = (_Float16*)(ws + 0);         //  4,194,304 B  (32768 x 64)
    int*      idx    = (int*)(ws + 4194304);        //  1,398,784 B
    float*    fsub   = (float*)(ws + 5593088);      //  5,595,136 B
    _Float16* afp    = (_Float16*)(ws + 11188224);  // 12,582,912 B  (32768 x 192)
    _Float16* zbuf   = (_Float16*)(ws + 23771136);  //  6,291,456 B  (32768 x 96)
    float*    zc     = (float*)(ws + 30062592);     //  8,388,608 B
    float*    stats  = (float*)(ws + 38451200);     //        512 B
    _Float16* wsa_t  = (_Float16*)(ws + 38451712);  //     24,576 B  (128 x 96)
    _Float16* wfp_t  = (_Float16*)(ws + 38476288);  //     24,576 B  (64 x 192)
    _Float16* wfin_t = (_Float16*)(ws + 38500864);  //     12,288 B  (64 x 96)

    float* out  = (float*)d_out;
    float* outz = out + BB*NN*3;

    initk<<<(ROWS_PT*96/2 + 255)/256, 256, 0, stream>>>(data, out, fsub, (unsigned int*)zbuf);
    prep_w<<<(64*192 + 255)/256, 256, 0, stream>>>(W_sa, W_fp, W_fin, wsa_t, wfp_t, wfin_t);
    stemk<<<(ROWS_PT*64 + 255)/256, 256, 0, stream>>>(data, W_stem, b_stem, f0h, afp);
    ballq<<<(BB*MM + 63)/64, 64, 0, stream>>>(data, idx);
    sa_gemm<<<(ROWS_SA/16) * 8, 32, 0, stream>>>(data, f0h, idx, wsa_t, b_sa, (int*)fsub);
    knn3<<<(ROWS_PT + 63)/64, 64, 0, stream>>>(data, fsub, afp);
    fp_gemm<<<(ROWS_PT/16) * 4, 32, 0, stream>>>(afp, wfp_t, b_fp, zbuf);
    edk_kernel<<<(ROWS_PT + 63)/64, 64, 0, stream>>>(data, W_e1, b_e1, W_e2, b_e2, zbuf);
    fin_gemm<<<(ROWS_PT/16) * 4, 32, 0, stream>>>(zbuf, wfin_t, b_fin, zc);
    bnstats<<<64, 256, 0, stream>>>(zc, stats);
    finalize<<<(ROWS_PT*64 + 255)/256, 256, 0, stream>>>(zc, stats, gamma, beta, outz);
}
